// MambaHSIAutoencoder_23502061044158
// MI455X (gfx1250) — matrix operand order repaired and checked by reading, not compiled, not run on hardware
//
#include <hip/hip_runtime.h>
#include <hip/hip_bf16.h>

// MambaHSI autoencoder for MI455X (gfx1250, wave32).
// GEMMs (mamba in/x/out projections) run on v_wmma_f32_16x16x32_f16 (f32
// accumulate), with float4 (global_load_b128) fragment loads and multi-tile
// accumulators per wave (A-fragment reuse across 3-4 WMMAs). SSM scan is a
// register-resident VALU kernel (one block per sequence). Convs are direct
// VALU kernels (tiny fraction of the work).
// Workspace requirement: ~22.0M floats (~88 MB).

#define C_IN   224
#define DM     64
#define DI     128
#define DS     16
#define DTR    4
#define DCONV  4
#define LAT    16
#define NL     2
#define NB     2
#define H0     64
#define H1     32
#define H2     16
#define NSEQ   512              // NB*H2*H2 sequences
#define LSEQ   C_IN             // sequence length = 224
#define TTOK   (NSEQ * LSEQ)    // 114688 tokens
#define CHSEQ  128              // sequences per chunk
#define NCHUNK (NSEQ / CHSEQ)   // 4
#define CHT    (CHSEQ * LSEQ)   // 28672 tokens per chunk

typedef __attribute__((ext_vector_type(16))) _Float16 v16h;
typedef __attribute__((ext_vector_type(8)))  float    v8f;

// ---------------------------------------------------------------------------
// WMMA fragment loaders (CDNA5 wave32 layouts, f16 A/B, f32 C/D)
// A 16x32 (MxK): lane%16 = M, element e: k = e + 16*(lane>>4) -> one
//   contiguous 16-float run per lane (matches B). Loaded as 4x float4
//   (global_load_b128).
// B 32x16 (KxN): lane%16 = N, element e: k = e + 16*(lane>>4) -> one
//   contiguous 16-float run per lane. Loaded as 4x float4.
// C/D 16x16 f32: vgpr r: M = r + 8*(lane>>4), N = lane&15
// ---------------------------------------------------------------------------
__device__ __forceinline__ v16h load_a_f32(const float* __restrict__ A, int lda,
                                           int m0, int k0) {
    const int lane = threadIdx.x & 31;
    const float* row = A + (size_t)(m0 + (lane & 15)) * lda + k0 + ((lane >> 4) << 4);
    const float4 a0 = *reinterpret_cast<const float4*>(row + 0);
    const float4 a1 = *reinterpret_cast<const float4*>(row + 4);
    const float4 a2 = *reinterpret_cast<const float4*>(row + 8);
    const float4 a3 = *reinterpret_cast<const float4*>(row + 12);
    v16h a;
    a[0]  = (_Float16)a0.x; a[1]  = (_Float16)a0.y;
    a[2]  = (_Float16)a0.z; a[3]  = (_Float16)a0.w;
    a[4]  = (_Float16)a1.x; a[5]  = (_Float16)a1.y;
    a[6]  = (_Float16)a1.z; a[7]  = (_Float16)a1.w;
    a[8]  = (_Float16)a2.x; a[9]  = (_Float16)a2.y;
    a[10] = (_Float16)a2.z; a[11] = (_Float16)a2.w;
    a[12] = (_Float16)a3.x; a[13] = (_Float16)a3.y;
    a[14] = (_Float16)a3.z; a[15] = (_Float16)a3.w;
    return a;
}

// B = W^T where W is (N,K) row-major. Pads n >= Nmax with zeros.
__device__ __forceinline__ v16h load_b_wT(const float* __restrict__ W, int ldk,
                                          int n0, int k0, int Nmax) {
    const int lane = threadIdx.x & 31;
    const int n = n0 + (lane & 15);
    v16h b;
    if (n < Nmax) {
        const float* row = W + (size_t)n * ldk + k0 + ((lane >> 4) << 4);
        const float4 b0 = *reinterpret_cast<const float4*>(row + 0);
        const float4 b1 = *reinterpret_cast<const float4*>(row + 4);
        const float4 b2 = *reinterpret_cast<const float4*>(row + 8);
        const float4 b3 = *reinterpret_cast<const float4*>(row + 12);
        b[0]  = (_Float16)b0.x; b[1]  = (_Float16)b0.y;
        b[2]  = (_Float16)b0.z; b[3]  = (_Float16)b0.w;
        b[4]  = (_Float16)b1.x; b[5]  = (_Float16)b1.y;
        b[6]  = (_Float16)b1.z; b[7]  = (_Float16)b1.w;
        b[8]  = (_Float16)b2.x; b[9]  = (_Float16)b2.y;
        b[10] = (_Float16)b2.z; b[11] = (_Float16)b2.w;
        b[12] = (_Float16)b3.x; b[13] = (_Float16)b3.y;
        b[14] = (_Float16)b3.z; b[15] = (_Float16)b3.w;
    } else {
#pragma unroll
        for (int e = 0; e < 16; ++e) b[e] = (_Float16)0.0f;
    }
    return b;
}

// ---------------------------------------------------------------------------
// Encoder convs
// ---------------------------------------------------------------------------
__global__ void k_conv1(const float* __restrict__ x, const float* __restrict__ w,
                        const float* __restrict__ b, float* __restrict__ out) {
    int idx = blockIdx.x * blockDim.x + threadIdx.x;
    const int total = NB * 128 * H1 * H1;
    if (idx >= total) return;
    const int ow = idx % H1, oh = (idx / H1) % H1;
    const int co = (idx / (H1 * H1)) % 128, n = idx / (128 * H1 * H1);
    float acc = b[co];
    for (int ci = 0; ci < C_IN; ++ci)
#pragma unroll
        for (int kh = 0; kh < 4; ++kh) {
            const int ih = oh * 2 - 1 + kh;
            if ((unsigned)ih >= H0) continue;
#pragma unroll
            for (int kw = 0; kw < 4; ++kw) {
                const int iw = ow * 2 - 1 + kw;
                if ((unsigned)iw >= H0) continue;
                acc += x[((size_t)(n * C_IN + ci) * H0 + ih) * H0 + iw] *
                       w[((co * C_IN + ci) * 4 + kh) * 4 + kw];
            }
        }
    out[idx] = fmaxf(acc, 0.f);
}

__global__ void k_conv2(const float* __restrict__ h1, const float* __restrict__ w,
                        const float* __restrict__ b, float* __restrict__ ss) {
    int idx = blockIdx.x * blockDim.x + threadIdx.x;
    const int total = NB * C_IN * H2 * H2;
    if (idx >= total) return;
    const int ow = idx % H2, oh = (idx / H2) % H2;
    const int co = (idx / (H2 * H2)) % C_IN, n = idx / (C_IN * H2 * H2);
    float acc = b[co];
    for (int ci = 0; ci < 128; ++ci)
#pragma unroll
        for (int kh = 0; kh < 4; ++kh) {
            const int ih = oh * 2 - 1 + kh;
            if ((unsigned)ih >= H1) continue;
#pragma unroll
            for (int kw = 0; kw < 4; ++kw) {
                const int iw = ow * 2 - 1 + kw;
                if ((unsigned)iw >= H1) continue;
                acc += h1[((size_t)(n * 128 + ci) * H1 + ih) * H1 + iw] *
                       w[((co * 128 + ci) * 4 + kh) * 4 + kw];
            }
        }
    const int pix = (n * H2 + oh) * H2 + ow;   // transpose(0,2,3,1) pixel order
    ss[(size_t)pix * C_IN + co] = fmaxf(acc, 0.f);
}

// s[t, d] = ss[t] * w[d] + b[d]   (rank-1 in-projection, inproj_w is (64,1))
__global__ void k_expand(const float* __restrict__ ss, const float* __restrict__ w,
                         const float* __restrict__ b, float* __restrict__ S) {
    int idx = blockIdx.x * blockDim.x + threadIdx.x;
    if (idx >= TTOK * DM) return;
    const int d = idx % DM;
    const int t = idx / DM;
    S[idx] = ss[t] * w[d] + b[d];
}

// ---------------------------------------------------------------------------
// Mamba building blocks
// ---------------------------------------------------------------------------
__global__ void k_ln(const float* __restrict__ S, const float* __restrict__ g,
                     const float* __restrict__ bta, float* __restrict__ T, int ntok) {
    int t = blockIdx.x * blockDim.x + threadIdx.x;
    if (t >= ntok) return;
    const float* x = S + (size_t)t * DM;
    float m = 0.f;
#pragma unroll
    for (int i = 0; i < DM; ++i) m += x[i];
    m *= (1.0f / DM);
    float v = 0.f;
#pragma unroll
    for (int i = 0; i < DM; ++i) { const float d = x[i] - m; v += d * d; }
    const float r = rsqrtf(v * (1.0f / DM) + 1e-5f);
    float* o = T + (size_t)t * DM;
#pragma unroll
    for (int i = 0; i < DM; ++i) o[i] = (x[i] - m) * r * g[i] + bta[i];
}

// xz = tln @ in_w^T ; cols [0,128) -> xm, [128,256) -> z.
// One wave computes a 16(M) x 64(N) strip: A fragment loaded once per k-step,
// reused by 4 WMMAs. The xm/z split is on a tile boundary -> uniform select.
__global__ void k_gemm_inproj(const float* __restrict__ Aact, const float* __restrict__ W,
                              float* __restrict__ xm, float* __restrict__ zb, int Mtiles) {
    const int wid = (blockIdx.x * blockDim.x + threadIdx.x) >> 5;
    if (wid >= Mtiles * 4) return;
    const int mt = wid >> 2, ng = wid & 3;       // ng: 64-column group
    const int lane = threadIdx.x & 31;
    v8f acc[4] = {};
#pragma unroll
    for (int kb = 0; kb < DM; kb += 32) {
        const v16h a = load_a_f32(Aact, DM, mt * 16, kb);
#pragma unroll
        for (int j = 0; j < 4; ++j) {
            const v16h b = load_b_wT(W, DM, ng * 64 + j * 16, kb, 2 * DI);
            acc[j] = __builtin_amdgcn_wmma_f32_16x16x32_f16(false, a, false, b,
                                                            (short)0, acc[j], false, false);
        }
    }
    const int mbase = mt * 16 + ((lane >> 4) << 3);
#pragma unroll
    for (int j = 0; j < 4; ++j) {
        const int ncol = ng * 64 + j * 16 + (lane & 15);
        float* dst = (ncol < DI) ? (xm + ncol) : (zb + (ncol - DI));
#pragma unroll
        for (int r = 0; r < 8; ++r) dst[(size_t)(mbase + r) * DI] = acc[j][r];
    }
}

// depthwise causal conv1d (k=4, left pad 3) + SiLU
__global__ void k_dwconv(const float* __restrict__ xm, const float* __restrict__ cw,
                         const float* __restrict__ cb, float* __restrict__ xmc, int ntok) {
    int idx = blockIdx.x * blockDim.x + threadIdx.x;
    if (idx >= ntok * DI) return;
    const int d = idx % DI;
    const int t = idx / DI;
    const int l = t % LSEQ;
    float acc = cb[d];
#pragma unroll
    for (int j = 0; j < DCONV; ++j) {
        const int lj = l - (DCONV - 1) + j;
        if (lj >= 0)
            acc += xm[(size_t)(t - (DCONV - 1) + j) * DI + d] * cw[d * DCONV + j];
    }
    acc = acc / (1.f + __expf(-acc));   // silu
    xmc[(size_t)t * DI + d] = acc;
}

// dbl = xmc @ x_w^T  (N=36 over 3 tiles). One wave per M-tile, full N strip.
__global__ void k_gemm_xproj(const float* __restrict__ Aact, const float* __restrict__ W,
                             float* __restrict__ dbl, int Mtiles) {
    const int wid = (blockIdx.x * blockDim.x + threadIdx.x) >> 5;
    if (wid >= Mtiles) return;
    const int mt = wid;
    const int lane = threadIdx.x & 31;
    v8f acc[3] = {};
#pragma unroll
    for (int kb = 0; kb < DI; kb += 32) {
        const v16h a = load_a_f32(Aact, DI, mt * 16, kb);
#pragma unroll
        for (int j = 0; j < 3; ++j) {
            const v16h b = load_b_wT(W, DI, j * 16, kb, DTR + 2 * DS);
            acc[j] = __builtin_amdgcn_wmma_f32_16x16x32_f16(false, a, false, b,
                                                            (short)0, acc[j], false, false);
        }
    }
    const int mbase = mt * 16 + ((lane >> 4) << 3);
#pragma unroll
    for (int j = 0; j < 3; ++j) {
        const int n = j * 16 + (lane & 15);
        if (n < DTR + 2 * DS) {
#pragma unroll
            for (int r = 0; r < 8; ++r)
                dbl[(size_t)(mbase + r) * (DTR + 2 * DS) + n] = acc[j][r];
        }
    }
}

// Selective scan: one block per sequence, thread = d_inner channel.
// Fuses dt_proj + softplus, state update, y reduction, +x*D, *silu(z).
// Overwrites xmc in place with the gated output y.
__global__ void k_scan(const float* __restrict__ dbl, const float* __restrict__ zb,
                       float* __restrict__ xmc,
                       const float* __restrict__ A_log, const float* __restrict__ dtw,
                       const float* __restrict__ dtb, const float* __restrict__ Dp) {
    const int d = threadIdx.x;              // 0..127
    const size_t base = (size_t)blockIdx.x * LSEQ;
    __shared__ float lB[DS], lC[DS], lR[DTR];
    float Arow[DS];
#pragma unroll
    for (int s = 0; s < DS; ++s) Arow[s] = -__expf(A_log[d * DS + s]);
    const float w0 = dtw[d * DTR + 0], w1 = dtw[d * DTR + 1];
    const float w2 = dtw[d * DTR + 2], w3 = dtw[d * DTR + 3];
    const float bdt = dtb[d], Dd = Dp[d];
    float h[DS];
#pragma unroll
    for (int s = 0; s < DS; ++s) h[s] = 0.f;
    for (int l = 0; l < LSEQ; ++l) {
        const size_t t = base + l;
        if (d < DS)                lB[d] = dbl[t * 36 + DTR + d];
        else if (d < 2 * DS)       lC[d - DS] = dbl[t * 36 + DTR + DS + (d - DS)];
        else if (d < 2 * DS + DTR) lR[d - 2 * DS] = dbl[t * 36 + (d - 2 * DS)];
        __syncthreads();
        float dtv = bdt + w0 * lR[0] + w1 * lR[1] + w2 * lR[2] + w3 * lR[3];
        dtv = (dtv > 20.f) ? dtv : __logf(1.f + __expf(dtv));   // softplus
        const float xv = xmc[t * DI + d];
        const float zv = zb[t * DI + d];
        const float dx = dtv * xv;
        float y = 0.f;
#pragma unroll
        for (int s = 0; s < DS; ++s) {
            const float dA = __expf(dtv * Arow[s]);
            h[s] = dA * h[s] + dx * lB[s];
            y += h[s] * lC[s];
        }
        y = (y + xv * Dd) * (zv / (1.f + __expf(-zv)));
        xmc[t * DI + d] = y;
        __syncthreads();
    }
}

// s += y @ out_w^T : one wave per M-tile computes the full N=64 strip,
// accumulating into the residual stream via the C operand.
__global__ void k_gemm_outproj(const float* __restrict__ Y, const float* __restrict__ W,
                               float* __restrict__ Sres, int Mtiles) {
    const int wid = (blockIdx.x * blockDim.x + threadIdx.x) >> 5;
    if (wid >= Mtiles) return;
    const int mt = wid;
    const int lane = threadIdx.x & 31;
    const int mbase = mt * 16 + ((lane >> 4) << 3);
    v8f acc[4];
#pragma unroll
    for (int j = 0; j < 4; ++j) {
        const int n = j * 16 + (lane & 15);
#pragma unroll
        for (int r = 0; r < 8; ++r) acc[j][r] = Sres[(size_t)(mbase + r) * DM + n];
    }
#pragma unroll
    for (int kb = 0; kb < DI; kb += 32) {
        const v16h a = load_a_f32(Y, DI, mt * 16, kb);
#pragma unroll
        for (int j = 0; j < 4; ++j) {
            const v16h b = load_b_wT(W, DI, j * 16, kb, DM);
            acc[j] = __builtin_amdgcn_wmma_f32_16x16x32_f16(false, a, false, b,
                                                            (short)0, acc[j], false, false);
        }
    }
#pragma unroll
    for (int j = 0; j < 4; ++j) {
        const int n = j * 16 + (lane & 15);
#pragma unroll
        for (int r = 0; r < 8; ++r) Sres[(size_t)(mbase + r) * DM + n] = acc[j][r];
    }
}

// ---------------------------------------------------------------------------
// Decoder
// ---------------------------------------------------------------------------
__global__ void k_head(const float* __restrict__ S, const float* __restrict__ ow,
                       const float* __restrict__ ob, float* __restrict__ feat) {
    int t = blockIdx.x * blockDim.x + threadIdx.x;
    if (t >= TTOK) return;
    float acc = ob[0];
#pragma unroll
    for (int d = 0; d < DM; ++d) acc += S[(size_t)t * DM + d] * ow[d];
    const int c = t % C_IN;
    const int pix = t / C_IN;
    const int wq = pix % H2, hq = (pix / H2) % H2, n = pix / (H2 * H2);
    feat[((size_t)(n * C_IN + c) * H2 + hq) * H2 + wq] = acc;
}

__global__ void k_cred(const float* __restrict__ feat, const float* __restrict__ w,
                       const float* __restrict__ b, float* __restrict__ z2) {
    int idx = blockIdx.x * blockDim.x + threadIdx.x;
    const int total = NB * LAT * H2 * H2;
    if (idx >= total) return;
    const int wq = idx % H2, hq = (idx / H2) % H2;
    const int lat = (idx / (H2 * H2)) % LAT, n = idx / (LAT * H2 * H2);
    float acc = b[lat];
    for (int c = 0; c < C_IN; ++c)
        acc += feat[((size_t)(n * C_IN + c) * H2 + hq) * H2 + wq] * w[lat * C_IN + c];
    z2[idx] = acc;
}

// ConvTranspose2d(k=4,s=2,p=1): ih = (oh+1-kh)/2 when parity matches.
__global__ void k_dec1(const float* __restrict__ z2, const float* __restrict__ w,
                       const float* __restrict__ b, float* __restrict__ d1) {
    int idx = blockIdx.x * blockDim.x + threadIdx.x;
    const int total = NB * 128 * H1 * H1;
    if (idx >= total) return;
    const int ow = idx % H1, oh = (idx / H1) % H1;
    const int co = (idx / (H1 * H1)) % 128, n = idx / (128 * H1 * H1);
    float acc = b[co];
#pragma unroll
    for (int kh = 0; kh < 4; ++kh) {
        const int th = oh + 1 - kh;
        if (th & 1) continue;
        const int ih = th >> 1;
        if ((unsigned)ih >= H2) continue;
#pragma unroll
        for (int kw = 0; kw < 4; ++kw) {
            const int tw = ow + 1 - kw;
            if (tw & 1) continue;
            const int iw = tw >> 1;
            if ((unsigned)iw >= H2) continue;
            for (int ci = 0; ci < LAT; ++ci)
                acc += z2[((size_t)(n * LAT + ci) * H2 + ih) * H2 + iw] *
                       w[((ci * 128 + co) * 4 + kh) * 4 + kw];
        }
    }
    d1[idx] = fmaxf(acc, 0.f);
}

__global__ void k_dec2(const float* __restrict__ d1, const float* __restrict__ w,
                       const float* __restrict__ b, float* __restrict__ out) {
    int idx = blockIdx.x * blockDim.x + threadIdx.x;
    const int total = NB * C_IN * H0 * H0;
    if (idx >= total) return;
    const int ow = idx % H0, oh = (idx / H0) % H0;
    const int co = (idx / (H0 * H0)) % C_IN, n = idx / (C_IN * H0 * H0);
    float acc = b[co];
#pragma unroll
    for (int kh = 0; kh < 4; ++kh) {
        const int th = oh + 1 - kh;
        if (th & 1) continue;
        const int ih = th >> 1;
        if ((unsigned)ih >= H1) continue;
#pragma unroll
        for (int kw = 0; kw < 4; ++kw) {
            const int tw = ow + 1 - kw;
            if (tw & 1) continue;
            const int iw = tw >> 1;
            if ((unsigned)iw >= H1) continue;
            for (int ci = 0; ci < 128; ++ci)
                acc += d1[((size_t)(n * 128 + ci) * H1 + ih) * H1 + iw] *
                       w[((ci * C_IN + co) * 4 + kh) * 4 + kw];
        }
    }
    out[idx] = acc;
}

// ---------------------------------------------------------------------------
// Launcher
// ---------------------------------------------------------------------------
extern "C" void kernel_launch(void* const* d_in, const int* in_sizes, int n_in,
                              void* d_out, int out_size, void* d_ws, size_t ws_size,
                              hipStream_t stream) {
    const float* x        = (const float*)d_in[0];
    const float* sconv1_w = (const float*)d_in[1];
    const float* sconv1_b = (const float*)d_in[2];
    const float* sconv2_w = (const float*)d_in[3];
    const float* sconv2_b = (const float*)d_in[4];
    const float* inproj_w = (const float*)d_in[5];
    const float* inproj_b = (const float*)d_in[6];
    const float* ln_g     = (const float*)d_in[7];
    const float* ln_b     = (const float*)d_in[8];
    const float* m_in_w   = (const float*)d_in[9];
    const float* m_conv_w = (const float*)d_in[10];
    const float* m_conv_b = (const float*)d_in[11];
    const float* m_x_w    = (const float*)d_in[12];
    const float* m_dt_w   = (const float*)d_in[13];
    const float* m_dt_b   = (const float*)d_in[14];
    const float* m_A_log  = (const float*)d_in[15];
    const float* m_D      = (const float*)d_in[16];
    const float* m_out_w  = (const float*)d_in[17];
    const float* outp_w   = (const float*)d_in[18];
    const float* outp_b   = (const float*)d_in[19];
    const float* cred_w   = (const float*)d_in[20];
    const float* cred_b   = (const float*)d_in[21];
    const float* dec1_w   = (const float*)d_in[22];
    const float* dec1_b   = (const float*)d_in[23];
    const float* dec2_w   = (const float*)d_in[24];
    const float* dec2_b   = (const float*)d_in[25];

    float* ws = (float*)d_ws;
    // workspace layout (floats); total ~21.98M floats (~88 MB)
    float* h1   = ws;                              // 262144
    float* ss   = h1   + 262144;                   // 114688
    float* S    = ss   + 114688;                   // 7340032
    float* tln  = S    + 7340032;                  // 1835008
    float* xm   = tln  + 1835008;                  // 3670016
    float* xmc  = xm   + 3670016;                  // 3670016
    float* zbuf = xmc  + 3670016;                  // 3670016
    float* dblb = zbuf + 3670016;                  // 1032192
    float* feat = dblb + 1032192;                  // 114688
    float* z2   = feat + 114688;                   // 8192
    float* d1   = z2   + 8192;                     // 262144

    const int BLK = 256;
    k_conv1<<<(NB * 128 * H1 * H1 + BLK - 1) / BLK, BLK, 0, stream>>>(x, sconv1_w, sconv1_b, h1);
    k_conv2<<<(NB * C_IN * H2 * H2 + BLK - 1) / BLK, BLK, 0, stream>>>(h1, sconv2_w, sconv2_b, ss);
    k_expand<<<(TTOK * DM + BLK - 1) / BLK, BLK, 0, stream>>>(ss, inproj_w, inproj_b, S);

    const int Mt = CHT / 16;  // 1792 M-tiles per chunk
    for (int i = 0; i < NL; ++i) {
        const float* Wi = m_in_w  + (size_t)i * 2 * DI * DM;
        const float* cw = m_conv_w + (size_t)i * DI * DCONV;
        const float* cb = m_conv_b + (size_t)i * DI;
        const float* Wx = m_x_w   + (size_t)i * (DTR + 2 * DS) * DI;
        const float* dw = m_dt_w  + (size_t)i * DI * DTR;
        const float* db = m_dt_b  + (size_t)i * DI;
        const float* Al = m_A_log + (size_t)i * DI * DS;
        const float* Dp = m_D     + (size_t)i * DI;
        const float* Wo = m_out_w + (size_t)i * DM * DI;
        const float* g  = ln_g + (size_t)i * DM;
        const float* bb = ln_b + (size_t)i * DM;
        for (int c = 0; c < NCHUNK; ++c) {
            float* Sc = S + (size_t)c * CHT * DM;
            k_ln<<<(CHT + BLK - 1) / BLK, BLK, 0, stream>>>(Sc, g, bb, tln, CHT);
            k_gemm_inproj<<<(Mt * 4 * 32 + BLK - 1) / BLK, BLK, 0, stream>>>(tln, Wi, xm, zbuf, Mt);
            k_dwconv<<<(CHT * DI + BLK - 1) / BLK, BLK, 0, stream>>>(xm, cw, cb, xmc, CHT);
            k_gemm_xproj<<<(Mt * 32 + BLK - 1) / BLK, BLK, 0, stream>>>(xmc, Wx, dblb, Mt);
            k_scan<<<CHSEQ, DI, 0, stream>>>(dblb, zbuf, xmc, Al, dw, db, Dp);
            k_gemm_outproj<<<(Mt * 32 + BLK - 1) / BLK, BLK, 0, stream>>>(xmc, Wo, Sc, Mt);
        }
    }

    k_head<<<(TTOK + BLK - 1) / BLK, BLK, 0, stream>>>(S, outp_w, outp_b, feat);
    k_cred<<<(NB * LAT * H2 * H2 + BLK - 1) / BLK, BLK, 0, stream>>>(feat, cred_w, cred_b, z2);
    k_dec1<<<(NB * 128 * H1 * H1 + BLK - 1) / BLK, BLK, 0, stream>>>(z2, dec1_w, dec1_b, d1);
    k_dec2<<<(NB * C_IN * H0 * H0 + BLK - 1) / BLK, BLK, 0, stream>>>(d1, dec2_w, dec2_b, (float*)d_out);
}
